// QuantLinearORT_31035433681260
// MI455X (gfx1250) — matrix operand order repaired and checked by reading, compiled, not run on hardware
//
#include <hip/hip_runtime.h>
#include <stdint.h>

typedef __attribute__((ext_vector_type(16))) _Float16     v16h;
typedef __attribute__((ext_vector_type(2)))  _Float16     v2h;
typedef __attribute__((ext_vector_type(8)))  float        v8f;
typedef __attribute__((ext_vector_type(4)))  unsigned int v4u;
typedef __attribute__((ext_vector_type(8)))  int          v8i;
typedef __attribute__((ext_vector_type(4)))  int          v4i;

#define M_TOT 4096
#define N_TOT 11008
#define K_TOT 4096
#define QR    32
#define KW    64          // packed int32 words per (row, group)
#define BM    128
#define BN    128
#define BK    64
#define STEPS (K_TOT / BK)   // 64 K-steps

#if defined(__has_builtin)
#if __has_builtin(__builtin_amdgcn_tensor_load_to_lds)
#define HAVE_TDM 1
#endif
#endif
#ifndef HAVE_TDM
#define HAVE_TDM 0
#endif

union FragU { v16h v; uint32_t u[8]; };

// exact packed dequant: 0x6400|q = f16(1024+q); (1024+q)-(1024+zp) exact (Sterbenz);
// one rounding from the v_pk_mul_f16 by the group scale.
__device__ __forceinline__ uint32_t dq_pair(uint32_t w, v2h bias2, v2h sc2) {
    uint32_t packed = 0x64006400u | (w & 15u) | ((w & 0xF0u) << 12);
    v2h hv = __builtin_bit_cast(v2h, packed);
    hv = (hv - bias2) * sc2;
    return __builtin_bit_cast(uint32_t, hv);
}

// ---------------- pre-pass: x fp32 -> f16 into workspace ----------------
__global__ __launch_bounds__(256)
void cvt_f32_to_f16(const float* __restrict__ x, _Float16* __restrict__ y, int n4) {
    int i = blockIdx.x * blockDim.x + threadIdx.x;
    const int stride = gridDim.x * blockDim.x;
    const float4* xv = (const float4*)x;
    for (; i < n4; i += stride) {
        float4 f = xv[i];
        v2h a0; a0[0] = (_Float16)f.x; a0[1] = (_Float16)f.y;
        v2h a1; a1[0] = (_Float16)f.z; a1[1] = (_Float16)f.w;
        uint32_t* dst = (uint32_t*)(y + (size_t)i * 4);
        dst[0] = __builtin_bit_cast(uint32_t, a0);
        dst[1] = __builtin_bit_cast(uint32_t, a1);
    }
}

// ---------------- main GEMM ----------------
// Wave w owns N-strip [blockN + 16w, +16) and all 8 M-tiles.
// B fragments dequantized straight into registers; x tile double-buffered in LDS
// and fed by per-wave TDM slices overlapped with compute; one barrier per K-step.
template <bool USE_WS>
__global__ __launch_bounds__(256)
void qlinear_w4_wmma(const float*    __restrict__ x,
                     const _Float16* __restrict__ xf16,
                     const int*      __restrict__ qweight,
                     const int*      __restrict__ qzeros,
                     const float*    __restrict__ scales,
                     float*          __restrict__ out)
{
    __shared__ _Float16 xs[2][BM][BK];    // double-buffered 128x64 f16 tile (32 KB)

    const int tid  = threadIdx.x;
    const int lane = tid & 31;
    const int wave = tid >> 5;            // 0..7 -> N-strip
    const int half = lane >> 4;           // 0/1: K-halves of B / row-halves of C
    const int r    = lane & 15;

    const int blockN = blockIdx.x * BN;
    const int blockM = blockIdx.y * BM;

    const int nCol = blockN + wave * 16 + r;          // this lane's output column
    const int* qrow = qweight + (size_t)nCol * (QR * KW);

    const int srow = tid >> 1;            // copy-path staging row
    const int skh  = tid & 1;

    v8f acc[8] = {};

    // stage activation tile for K-step s into xs[s&1]
    auto stage = [&](int s) {
        const int k0 = s * BK;
        if (USE_WS) {
#if HAVE_TDM
            // uniform across waves: each wave DMAs its own 16-row slice
            uint64_t ga = (uint64_t)(uintptr_t)
                (xf16 + (size_t)(blockM + wave * 16) * K_TOT + k0);
            uint32_t lds = (uint32_t)(uintptr_t)&xs[s & 1][wave * 16][0];
            v4u d0;
            d0[0] = 1u;                                  // count=1
            d0[1] = lds;                                 // lds_addr
            d0[2] = (uint32_t)ga;                        // global_addr lo
            d0[3] = (uint32_t)((ga >> 32) & 0x01FFFFFFu)
                  | (2u << 30);                          // type=2
            v8i d1;
            d1[0] = (int)(1u << 16);                     // data_size=2B
            d1[1] = (int)((uint32_t)(K_TOT & 0xFFFF) << 16);   // tensor_dim0 lo
            d1[2] = (int)(((uint32_t)K_TOT >> 16) & 0xFFFFu)
                  | (int)((uint32_t)(M_TOT & 0xFFFF) << 16);   // tensor_dim1 lo
            d1[3] = (int)((uint32_t)BK << 16);           // tile_dim0=64
            d1[4] = 16;                                  // tile_dim1=16
            d1[5] = (int)(uint32_t)K_TOT;                // tensor_dim0_stride
            d1[6] = 0;
            d1[7] = 0;
            v4i z4 = {0, 0, 0, 0};
#if __clang_major__ >= 23
            v8i z8 = {0, 0, 0, 0, 0, 0, 0, 0};
            __builtin_amdgcn_tensor_load_to_lds(d0, d1, z4, z4, z8, 0);
#else
            __builtin_amdgcn_tensor_load_to_lds(d0, d1, z4, z4, 0);
#endif
#else
            // plain f16 copy global -> LDS
            const uint4* src = (const uint4*)
                (xf16 + (size_t)(blockM + srow) * K_TOT + k0 + skh * 32);
            uint4* dst = (uint4*)&xs[s & 1][srow][skh * 32];
            #pragma unroll
            for (int j = 0; j < 4; ++j) dst[j] = src[j];
#endif
        } else {
            // fp32 load + convert + store
            const float4* src = (const float4*)
                (x + (size_t)(blockM + srow) * K_TOT + k0 + skh * 32);
            #pragma unroll
            for (int j = 0; j < 8; ++j) {
                float4 f = src[j];
                v2h a0; a0[0] = (_Float16)f.x; a0[1] = (_Float16)f.y;
                v2h a1; a1[0] = (_Float16)f.z; a1[1] = (_Float16)f.w;
                uint32_t* dst = (uint32_t*)&xs[s & 1][srow][skh * 32 + j * 4];
                dst[0] = __builtin_bit_cast(uint32_t, a0);
                dst[1] = __builtin_bit_cast(uint32_t, a1);
            }
        }
    };

    // packed q-words for K-step s: word offset = s*32 + half*8 (+0,4,16,20)
    auto loadQ = [&](int s, uint4 q[4]) {
        const int* p = qrow + s * 32 + half * 8;
        q[0] = *(const uint4*)(p);
        q[1] = *(const uint4*)(p + 4);
        q[2] = *(const uint4*)(p + 16);
        q[3] = *(const uint4*)(p + 20);
    };

    // ---- prologue: stage tile 0, fetch step-0 weights/params ----
    stage(0);
    uint4 qc[4];
    loadQ(0, qc);
    float scC = scales[nCol * QR];
    int   zwC = qzeros[nCol * 16];

    for (int s = 0; s < STEPS; ++s) {
        // prefetch next step's weight words + quant params into registers
        uint4 qn[4] = {};
        float scN = 0.0f;
        int   zwN = 0;
        if (s + 1 < STEPS) {
            loadQ(s + 1, qn);
            const int gN = (s + 1) >> 1;
            scN = scales[nCol * QR + gN];
            zwN = qzeros[nCol * 16 + (gN >> 1)];
        }
        if (s + 2 < STEPS)
            __builtin_prefetch(qrow + (s + 2) * 32 + half * 8, 0, 1);

        // ---- dequantize current step into B fragments (registers only) ----
        const int g  = s >> 1;
        const int zp = (zwC >> ((g & 1) * 4)) & 15;
        _Float16 hsc = (_Float16)scC;
        v2h sc2;   sc2[0] = hsc; sc2[1] = hsc;
        _Float16 hb = (_Float16)(float)(1024 + zp);      // exact in f16
        v2h bias2; bias2[0] = hb; bias2[1] = hb;

        FragU b0, b1;
        #pragma unroll
        for (int j = 0; j < 4; ++j) {
            const uint4 q = qc[j];
            FragU* bf = (j < 2) ? &b0 : &b1;
            const int o = (j & 1) * 4;
            bf->u[o + 0] = dq_pair(q.x, bias2, sc2);
            bf->u[o + 1] = dq_pair(q.y, bias2, sc2);
            bf->u[o + 2] = dq_pair(q.z, bias2, sc2);
            bf->u[o + 3] = dq_pair(q.w, bias2, sc2);
        }

#if HAVE_TDM
        if (USE_WS) __builtin_amdgcn_s_wait_tensorcnt(0);  // my slice of tile s done
#endif
        __syncthreads();             // all slices of tile s visible; buf (s+1)&1 free
        if (s + 1 < STEPS) stage(s + 1);   // DMA/copy next tile under compute

        // ---- 2 K-substeps x 8 M-tiles, B held in registers ----
        const _Float16 (*xb)[BK] = xs[s & 1];
        #pragma unroll
        for (int ks = 0; ks < 2; ++ks) {
            const FragU* bb = ks ? &b1 : &b0;
            #pragma unroll
            for (int mt = 0; mt < 8; ++mt) {
                FragU a;
                #pragma unroll
                for (int v = 0; v < 8; ++v) {
                    int kb = 2 * v + half * 16 + ks * 32;
                    a.u[v] = *(const uint32_t*)&xb[mt * 16 + r][kb];
                }
                acc[mt] = __builtin_amdgcn_wmma_f32_16x16x32_f16(
                    false, a.v, false, bb->v, (short)0, acc[mt], false, false);
            }
        }

        qc[0] = qn[0]; qc[1] = qn[1]; qc[2] = qn[2]; qc[3] = qn[3];
        scC = scN; zwC = zwN;
    }

    // ---- store C: lane's column nCol; rows mt*16 + i + 8*half (NT stream) ----
    float* ocol = out + nCol;
    #pragma unroll
    for (int mt = 0; mt < 8; ++mt) {
        const size_t mRow = (size_t)(blockM + mt * 16 + half * 8);
        #pragma unroll
        for (int i = 0; i < 8; ++i)
            __builtin_nontemporal_store(acc[mt][i], ocol + (mRow + i) * N_TOT);
    }
}

extern "C" void kernel_launch(void* const* d_in, const int* in_sizes, int n_in,
                              void* d_out, int out_size, void* d_ws, size_t ws_size,
                              hipStream_t stream) {
    (void)in_sizes; (void)n_in; (void)out_size;
    const float* x       = (const float*)d_in[0];
    const int*   qweight = (const int*)  d_in[1];
    const int*   qzeros  = (const int*)  d_in[2];
    const float* scales  = (const float*)d_in[3];
    float*       out     = (float*)d_out;

    dim3 grid(N_TOT / BN, M_TOT / BM);   // 86 x 32
    const size_t need = (size_t)M_TOT * K_TOT * sizeof(_Float16);  // 33.5 MB

    if (d_ws != nullptr && ws_size >= need) {
        _Float16* xf16 = (_Float16*)d_ws;
        cvt_f32_to_f16<<<4096, 256, 0, stream>>>(x, xf16, (M_TOT * K_TOT) / 4);
        qlinear_w4_wmma<true><<<grid, 256, 0, stream>>>(x, xf16, qweight, qzeros,
                                                        scales, out);
    } else {
        qlinear_w4_wmma<false><<<grid, 256, 0, stream>>>(x, nullptr, qweight, qzeros,
                                                         scales, out);
    }
}
